// RecurrentSpikLinearLayer_24283745092267
// MI455X (gfx1250) — matrix operand order repaired and checked by reading, not compiled, not run on hardware
//
#include <hip/hip_runtime.h>

typedef __attribute__((ext_vector_type(16))) __bf16 v16bf;
typedef __attribute__((ext_vector_type(8)))  __bf16 v8bf;
typedef __attribute__((ext_vector_type(8)))  float  v8f;

#define T_STEPS 4
#define D_K     512           // D_IN
#define D_N     512           // D_OUT
#define M_TOT   65536         // T*B*N = 4*64*256
#define SLICE   8388608       // B*N*D_OUT elements per time step
#define BM      128
#define BN      128
#define BK      32
#define LDA     40            // 32 + 8 pad (bf16 elems), 80 B row stride (16B aligned)
#define LDB     40

static __device__ __forceinline__ unsigned short f2bf(float f) {
    unsigned int u = __builtin_bit_cast(unsigned int, f);
    // round-to-nearest-even to bf16
    u += 0x7FFFu + ((u >> 16) & 1u);
    return (unsigned short)(u >> 16);
}
static __device__ __forceinline__ unsigned int pack2(float a, float b) {
    return (unsigned int)f2bf(a) | ((unsigned int)f2bf(b) << 16);
}

// ---------------------------------------------------------------------------
// Kernel 0: zero the BN stat accumulators (sum, sumsq = 1024 floats)
// ---------------------------------------------------------------------------
__global__ void zero_stats_kernel(float* __restrict__ stats) {
    int t = threadIdx.x;
    stats[t] = 0.0f; stats[t + 256] = 0.0f; stats[t + 512] = 0.0f; stats[t + 768] = 0.0f;
}

// ---------------------------------------------------------------------------
// Kernel 1: z = x @ W^T  (bf16 WMMA, fp32 accumulate), fused per-channel
//           sum / sumsq for BatchNorm. z written to d_out (scratch reuse).
// ---------------------------------------------------------------------------
__global__ __launch_bounds__(256) void gemm_stats_kernel(
        const float* __restrict__ x,      // [M_TOT, D_K]
        const float* __restrict__ W,      // [D_N, D_K]
        float* __restrict__ z,            // [M_TOT, D_N]  (== d_out, scratch)
        float* __restrict__ gSum,         // [D_N]
        float* __restrict__ gSq)          // [D_N]
{
    __shared__ unsigned short ldsA[BM * LDA];
    __shared__ unsigned short ldsB[BN * LDB];
    __shared__ float ldsSum[BN];
    __shared__ float ldsSq[BN];

    const int tid    = threadIdx.x;
    const int bx     = blockIdx.x;
    const int nTile  = bx & 3;            // 4 consecutive blocks share one x strip (L2 reuse)
    const int mTile  = bx >> 2;
    const size_t mBase = (size_t)mTile * BM;
    const int    nBase = nTile * BN;

    if (tid < BN) { ldsSum[tid] = 0.0f; ldsSq[tid] = 0.0f; }

    const int wave   = tid >> 5;
    const int lane   = tid & 31;
    const int lane16 = lane & 15;
    const int hl     = lane >> 4;         // 0 / 1 lane-half
    const int wm     = wave >> 2;         // 0..1  -> M offset wm*64
    const int wn     = wave & 3;          // 0..3  -> N offset wn*32

    // staging assignment: thread -> (row, half-of-32-floats)
    const int sr = tid >> 1;
    const int sh = tid & 1;

    v8f acc[4][2];
    #pragma unroll
    for (int mi = 0; mi < 4; ++mi)
        #pragma unroll
        for (int ni = 0; ni < 2; ++ni)
            acc[mi][ni] = {};

    for (int k0 = 0; k0 < D_K; k0 += BK) {
        __syncthreads();

        // ---- stage A (x) tile: 128x32 fp32 -> bf16 LDS ----
        {
            const float*  ap = x + (mBase + (size_t)sr) * D_K + k0 + sh * 16;
            const float4* a4 = (const float4*)ap;
            if (k0 + BK < D_K) __builtin_prefetch(ap + BK, 0, 0);   // global_prefetch_b8
            float4 f0 = a4[0], f1 = a4[1], f2 = a4[2], f3 = a4[3];
            uint4 u0, u1;
            u0.x = pack2(f0.x, f0.y); u0.y = pack2(f0.z, f0.w);
            u0.z = pack2(f1.x, f1.y); u0.w = pack2(f1.z, f1.w);
            u1.x = pack2(f2.x, f2.y); u1.y = pack2(f2.z, f2.w);
            u1.z = pack2(f3.x, f3.y); u1.w = pack2(f3.z, f3.w);
            uint4* dst = (uint4*)&ldsA[sr * LDA + sh * 16];
            dst[0] = u0; dst[1] = u1;
        }
        // ---- stage B (W) tile as [n][k]: B[k,n] = W[n,k] ----
        {
            const float*  wp = W + (size_t)(nBase + sr) * D_K + k0 + sh * 16;
            const float4* w4 = (const float4*)wp;
            float4 f0 = w4[0], f1 = w4[1], f2 = w4[2], f3 = w4[3];
            uint4 u0, u1;
            u0.x = pack2(f0.x, f0.y); u0.y = pack2(f0.z, f0.w);
            u0.z = pack2(f1.x, f1.y); u0.w = pack2(f1.z, f1.w);
            u1.x = pack2(f2.x, f2.y); u1.y = pack2(f2.z, f2.w);
            u1.z = pack2(f3.x, f3.y); u1.w = pack2(f3.z, f3.w);
            uint4* dst = (uint4*)&ldsB[sr * LDB + sh * 16];
            dst[0] = u0; dst[1] = u1;
        }
        __syncthreads();

        // ---- load fragments per documented 16-bit VGPR layouts ----
        v16bf afr[4], bfr[2];
        #pragma unroll
        for (int mi = 0; mi < 4; ++mi) {
            // A 16x32: lane-half 0 -> K 0..7 & 16..23 ; half 1 -> K 8..15 & 24..31
            const int row  = wm * 64 + mi * 16 + lane16;
            const int koff = hl * 8;
            const v8bf lo = *(const v8bf*)&ldsA[row * LDA + koff];
            const v8bf hi = *(const v8bf*)&ldsA[row * LDA + koff + 16];
            afr[mi] = __builtin_shufflevector(lo, hi, 0,1,2,3,4,5,6,7,8,9,10,11,12,13,14,15);
        }
        #pragma unroll
        for (int ni = 0; ni < 2; ++ni) {
            // B 32x16: lane-half 0 -> K 0..7 & 16..23 ; half 1 -> K 8..15 & 24..31
            const int rowN = wn * 32 + ni * 16 + lane16;
            const int kb   = hl * 8;
            const v8bf lo = *(const v8bf*)&ldsB[rowN * LDB + kb];
            const v8bf hi = *(const v8bf*)&ldsB[rowN * LDB + kb + 16];
            bfr[ni] = __builtin_shufflevector(lo, hi, 0,1,2,3,4,5,6,7,8,9,10,11,12,13,14,15);
        }
        #pragma unroll
        for (int mi = 0; mi < 4; ++mi)
            #pragma unroll
            for (int ni = 0; ni < 2; ++ni)
                acc[mi][ni] = __builtin_amdgcn_wmma_f32_16x16x32_bf16(
                    /*neg_a=*/false, afr[mi], /*neg_b=*/false, bfr[ni],
                    /*c_mod=*/(short)0, acc[mi][ni],
                    /*reuse_a=*/false, /*reuse_b=*/false);
    }

    // ---- write z + accumulate per-channel stats ----
    #pragma unroll
    for (int ni = 0; ni < 2; ++ni) {
        float s = 0.0f, q = 0.0f;
        const int gcol = nBase + wn * 32 + ni * 16 + lane16;
        #pragma unroll
        for (int mi = 0; mi < 4; ++mi) {
            #pragma unroll
            for (int j = 0; j < 8; ++j) {
                const float v = acc[mi][ni][j];
                const size_t gm = mBase + (size_t)(wm * 64 + mi * 16 + hl * 8 + j);
                z[gm * D_N + gcol] = v;
                s += v; q += v * v;
            }
        }
        atomicAdd(&ldsSum[wn * 32 + ni * 16 + lane16], s);   // ds_add_f32
        atomicAdd(&ldsSq [wn * 32 + ni * 16 + lane16], q);
    }
    __syncthreads();
    if (tid < BN) {
        atomicAdd(&gSum[nBase + tid], ldsSum[tid]);          // global_atomic_add_f32
        atomicAdd(&gSq [nBase + tid], ldsSq [tid]);
    }
}

// ---------------------------------------------------------------------------
// Kernel 2: finalize BN scale/shift per channel
// ---------------------------------------------------------------------------
__global__ void bn_finalize_kernel(const float* __restrict__ gSum,
                                   const float* __restrict__ gSq,
                                   const float* __restrict__ gamma,
                                   const float* __restrict__ beta,
                                   float* __restrict__ scale,
                                   float* __restrict__ shift)
{
    const int o = blockIdx.x * blockDim.x + threadIdx.x;
    if (o >= D_N) return;
    const float invM = 1.0f / (float)M_TOT;
    const float mean = gSum[o] * invM;
    const float var  = gSq[o] * invM - mean * mean;
    const float inv  = rsqrtf(var + 1e-5f);
    const float sc   = gamma[o] * inv;
    scale[o] = sc;
    shift[o] = beta[o] - mean * sc;
}

// ---------------------------------------------------------------------------
// Kernel 3: BN affine + 4-step LIF over T, in-place z -> spikes
// ---------------------------------------------------------------------------
__global__ __launch_bounds__(256) void bn_lif_kernel(float* __restrict__ z,
                                                     const float* __restrict__ scale,
                                                     const float* __restrict__ shift)
{
    const size_t i    = (size_t)blockIdx.x * blockDim.x + threadIdx.x;
    const size_t base = i * 4;                       // offset within one time slice
    const int    o    = (int)(base & (D_N - 1));
    const float4 sc = *(const float4*)(scale + o);
    const float4 sh = *(const float4*)(shift + o);

    float v0 = 0.f, v1 = 0.f, v2 = 0.f, v3 = 0.f;
    #pragma unroll
    for (int t = 0; t < T_STEPS; ++t) {
        float4* p = (float4*)(z + (size_t)t * SLICE + base);
        float4 zt = *p;
        const float a0 = zt.x * sc.x + sh.x;
        const float a1 = zt.y * sc.y + sh.y;
        const float a2 = zt.z * sc.z + sh.z;
        const float a3 = zt.w * sc.w + sh.w;
        v0 += (a0 - v0) * 0.5f;  v1 += (a1 - v1) * 0.5f;
        v2 += (a2 - v2) * 0.5f;  v3 += (a3 - v3) * 0.5f;
        const float s0 = (v0 >= 1.0f) ? 1.0f : 0.0f;
        const float s1 = (v1 >= 1.0f) ? 1.0f : 0.0f;
        const float s2 = (v2 >= 1.0f) ? 1.0f : 0.0f;
        const float s3 = (v3 >= 1.0f) ? 1.0f : 0.0f;
        v0 *= (1.0f - s0); v1 *= (1.0f - s1); v2 *= (1.0f - s2); v3 *= (1.0f - s3);
        float4 out; out.x = s0; out.y = s1; out.z = s2; out.w = s3;
        *p = out;
    }
}

// ---------------------------------------------------------------------------
extern "C" void kernel_launch(void* const* d_in, const int* in_sizes, int n_in,
                              void* d_out, int out_size, void* d_ws, size_t ws_size,
                              hipStream_t stream) {
    (void)in_sizes; (void)n_in; (void)out_size; (void)ws_size;
    const float* x     = (const float*)d_in[0];   // [4,64,256,512]
    const float* W     = (const float*)d_in[1];   // [512,512]
    const float* gamma = (const float*)d_in[2];   // [512]
    const float* beta  = (const float*)d_in[3];   // [512]
    float* out = (float*)d_out;                   // [4,64,256,512] — also z scratch
    float* ws  = (float*)d_ws;
    float* gSum  = ws;          // 512
    float* gSq   = ws + 512;    // 512
    float* scale = ws + 1024;   // 512
    float* shift = ws + 1536;   // 512

    zero_stats_kernel<<<1, 256, 0, stream>>>(gSum);
    gemm_stats_kernel<<<(M_TOT / BM) * (D_N / BN), 256, 0, stream>>>(x, W, out, gSum, gSq);
    bn_finalize_kernel<<<2, 256, 0, stream>>>(gSum, gSq, gamma, beta, scale, shift);
    bn_lif_kernel<<<SLICE / 4 / 256, 256, 0, stream>>>(out, scale, shift);
}
